// AttentionLayer_44023414784721
// MI455X (gfx1250) — matrix operand order repaired and checked by reading, not compiled, not run on hardware
//
#include <hip/hip_runtime.h>
#include <math.h>

// ---------------------------------------------------------------------------
// AttentionLayer for MI455X (gfx1250, wave32, WMMA, async-to-LDS copies).
//   q = x @ Wq^T + bq ; k = x @ Wk^T + bk ; v = x @ Wv^T + bv
//   scores = (q @ k^T) / sqrt(H) ; out = softmax_row(scores) * v (elementwise)
// fp32 inputs are converted once to f16; all four GEMMs run through
// v_wmma_f32_16x16x32_f16 with fp32 accumulation, f16 tiles staged in LDS
// via GLOBAL_LOAD_ASYNC_TO_LDS_B128 (double-buffered).
// ---------------------------------------------------------------------------

#define SEQ 4096
#define HID 4096

typedef __attribute__((ext_vector_type(16))) _Float16 v16h;
typedef __attribute__((ext_vector_type(8)))  _Float16 half8;   // 16 bytes
typedef __attribute__((ext_vector_type(4)))  _Float16 half4;   // 8 bytes
typedef __attribute__((ext_vector_type(8)))  float    v8f;
typedef __attribute__((ext_vector_type(4)))  int      int4v;   // 16 bytes

#define LDSK       32     // K-depth of an LDS tile (one WMMA K-step)
#define LDS_STRIDE 40     // 32 + 8 pad halves; row pitch 80B (16B aligned)
#define BM         128
#define BN         256
#define WM         64
#define WN         64

#if __has_builtin(__builtin_amdgcn_global_load_async_to_lds_b128)
#define HAVE_ASYNC_LDS 1
#else
#define HAVE_ASYNC_LDS 0
#endif

// 16-byte global -> LDS copy (async if available on this toolchain).
// Builtin signature (from hipcc diagnostic): first arg is int4 addrspace(1)*.
__device__ __forceinline__ void copy16_g2l(const _Float16* g, _Float16* l) {
#if HAVE_ASYNC_LDS
    __builtin_amdgcn_global_load_async_to_lds_b128(
        (__attribute__((address_space(1))) int4v*)g,
        (__attribute__((address_space(3))) int4v*)l,
        /*offset=*/0, /*cpol=*/0);
#else
    *(half8*)l = *(const half8*)g;
#endif
}

__device__ __forceinline__ void wait_copies() {
#if HAVE_ASYNC_LDS
#if __has_builtin(__builtin_amdgcn_s_wait_asynccnt)
    __builtin_amdgcn_s_wait_asynccnt(0);
#else
    asm volatile("s_wait_asynccnt 0x0" ::: "memory");
#endif
#endif
}

__device__ __forceinline__ v16h make_v16(half8 lo, half8 hi) {
    union { half8 h[2]; v16h v; } u;
    u.h[0] = lo; u.h[1] = hi;
    return u.v;
}

// A fragment, 16x32 f16 (MxK). Lane L: row M=L%16; lanes 0-15 carry K=0..7 &
// K=16..23; lanes 16-31 carry K=8..15 & K=24..31 (ISA 7.12.2).
__device__ __forceinline__ v16h frag_a(const _Float16* tile, int rowBase, int lane) {
    const int m = lane & 15;
    const int h = lane >> 4;
    const _Float16* p = tile + (rowBase + m) * LDS_STRIDE + h * 8;
    return make_v16(*(const half8*)(p), *(const half8*)(p + 16));
}

// B fragment, 32x16 f16 (KxN), stored as B^T row-major (row=N, col=K).
// Lane L: col N=L%16; lanes 0-15 carry K=0..7 & K=16..23; lanes 16-31 K=8..15 & K=24..31.
__device__ __forceinline__ v16h frag_b(const _Float16* tile, int colBase, int lane) {
    const int n = lane & 15;
    const int h = lane >> 4;
    const _Float16* p = tile + (colBase + n) * LDS_STRIDE + h * 8;
    return make_v16(*(const half8*)(p), *(const half8*)(p + 16));
}

// Cooperative copy of ROWS x 32 f16 tile (row-major, leading dim ld) into LDS.
template <int ROWS>
__device__ __forceinline__ void load_tile_f16(const _Float16* __restrict__ G, int ld,
                                              int row0, int k0, _Float16* tile, int tid) {
    constexpr int CHUNKS = ROWS * 4;         // 16B chunks (8 halves each)
#pragma unroll
    for (int i = 0; i < CHUNKS / 256; ++i) {
        const int linear = tid + i * 256;
        const int r = linear >> 2;
        const int c = linear & 3;
        copy16_g2l(G + (size_t)(row0 + r) * ld + k0 + c * 8,
                   tile + r * LDS_STRIDE + c * 8);
    }
}

// C[M,N] = (A[M,K] @ B[N,K]^T) * scale + bias[N]
// Block tile 128x256, BK=32, 256 threads = 8 waves (2x4), 64x64 per wave
// -> 16 v_wmma per K-step per wave. Double-buffered LDS (60 KB).
template <bool HAS_BIAS, bool HALF_OUT>
__global__ __launch_bounds__(256)
void gemm_nt_wmma(const _Float16* __restrict__ A, const _Float16* __restrict__ B,
                  const float* __restrict__ bias, void* __restrict__ Cout,
                  int M, int N, int K, float scale)
{
    __shared__ _Float16 As[2][BM * LDS_STRIDE];
    __shared__ _Float16 Bs[2][BN * LDS_STRIDE];

    const int tid   = threadIdx.x;
    const int lane  = tid & 31;
    const int wave  = tid >> 5;
    const int waveM = wave >> 2;             // 0..1
    const int waveN = wave & 3;              // 0..3

    const int blockRow = blockIdx.y * BM;
    const int blockCol = blockIdx.x * BN;
    const int wRow = waveM * WM;
    const int wCol = waveN * WN;

    v8f zero;
#pragma unroll
    for (int r = 0; r < 8; ++r) zero[r] = 0.0f;

    v8f acc[4][4];
#pragma unroll
    for (int i = 0; i < 4; ++i)
#pragma unroll
        for (int j = 0; j < 4; ++j) acc[i][j] = zero;

    // Prologue: stage first tiles.
    load_tile_f16<BM>(A, K, blockRow, 0, As[0], tid);
    load_tile_f16<BN>(B, K, blockCol, 0, Bs[0], tid);
    wait_copies();
    __syncthreads();

    int buf = 0;
    for (int k0 = 0; k0 < K; k0 += LDSK) {
        // Kick off next tile's copies while we compute on the current one.
        const int knext = k0 + LDSK;
        if (knext < K) {
            load_tile_f16<BM>(A, K, blockRow, knext, As[buf ^ 1], tid);
            load_tile_f16<BN>(B, K, blockCol, knext, Bs[buf ^ 1], tid);
        }

        v16h af[4];
        v16h bf[4];
#pragma unroll
        for (int i = 0; i < 4; ++i) af[i] = frag_a(As[buf], wRow + i * 16, lane);
#pragma unroll
        for (int j = 0; j < 4; ++j) bf[j] = frag_b(Bs[buf], wCol + j * 16, lane);

#pragma unroll
        for (int i = 0; i < 4; ++i)
#pragma unroll
            for (int j = 0; j < 4; ++j)
                acc[i][j] = __builtin_amdgcn_wmma_f32_16x16x32_f16(
                    false, af[i], false, bf[j],
                    (short)0, acc[i][j], false, false);

        wait_copies();
        __syncthreads();
        buf ^= 1;
    }

    // Epilogue. C/D layout: VGPR r, lane L -> M = r + (L>>4)*8, N = L&15.
    const int nLane = lane & 15;
    const int mHalf = (lane >> 4) * 8;
#pragma unroll
    for (int i = 0; i < 4; ++i) {
#pragma unroll
        for (int j = 0; j < 4; ++j) {
            const int col = blockCol + wCol + j * 16 + nLane;
            float bval = 0.0f;
            if constexpr (HAS_BIAS) bval = bias[col];
#pragma unroll
            for (int r = 0; r < 8; ++r) {
                const int row = blockRow + wRow + i * 16 + mHalf + r;
                const float val = acc[i][j][r] * scale + bval;
                if constexpr (HALF_OUT)
                    ((_Float16*)Cout)[(size_t)row * N + col] = (_Float16)val;
                else
                    ((float*)Cout)[(size_t)row * N + col] = val;
            }
        }
    }
}

// Elementwise fp32 -> f16 conversion (vectorized, one float4 per thread).
__global__ __launch_bounds__(256)
void cvt_f32_to_f16(const float* __restrict__ in, _Float16* __restrict__ out) {
    const size_t i = (size_t)blockIdx.x * 256 + threadIdx.x;
    const float4 f = ((const float4*)in)[i];
    half4 h;
    h.x = (_Float16)f.x; h.y = (_Float16)f.y;
    h.z = (_Float16)f.z; h.w = (_Float16)f.w;
    ((half4*)out)[i] = h;
}

// Fused row softmax + elementwise value mix, in place over `scores`.
__global__ __launch_bounds__(256)
void softmax_mix(float* __restrict__ scores, const float* __restrict__ v, int n)
{
    __shared__ float red[256];
    const int row = blockIdx.x;
    const int tid = threadIdx.x;
    float* srow = scores + (size_t)row * n;
    const float* vrow = v + (size_t)row * n;

    float4 s[4];
#pragma unroll
    for (int t = 0; t < 4; ++t) {
        const int c = (tid + t * 256) * 4;
        s[t] = *(const float4*)(srow + c);
    }

    float m = -INFINITY;
#pragma unroll
    for (int t = 0; t < 4; ++t)
        m = fmaxf(m, fmaxf(fmaxf(s[t].x, s[t].y), fmaxf(s[t].z, s[t].w)));
    red[tid] = m;
    __syncthreads();
    for (int off = 128; off > 0; off >>= 1) {
        if (tid < off) red[tid] = fmaxf(red[tid], red[tid + off]);
        __syncthreads();
    }
    const float rowMax = red[0];
    __syncthreads();

    float sum = 0.0f;
#pragma unroll
    for (int t = 0; t < 4; ++t) {
        s[t].x = __expf(s[t].x - rowMax);
        s[t].y = __expf(s[t].y - rowMax);
        s[t].z = __expf(s[t].z - rowMax);
        s[t].w = __expf(s[t].w - rowMax);
        sum += s[t].x + s[t].y + s[t].z + s[t].w;
    }
    red[tid] = sum;
    __syncthreads();
    for (int off = 128; off > 0; off >>= 1) {
        if (tid < off) red[tid] += red[tid + off];
        __syncthreads();
    }
    const float inv = 1.0f / red[0];

#pragma unroll
    for (int t = 0; t < 4; ++t) {
        const int c = (tid + t * 256) * 4;
        const float4 vv = *(const float4*)(vrow + c);
        float4 o;
        o.x = s[t].x * inv * vv.x;
        o.y = s[t].y * inv * vv.y;
        o.z = s[t].z * inv * vv.z;
        o.w = s[t].w * inv * vv.w;
        *(float4*)(srow + c) = o;
    }
}

extern "C" void kernel_launch(void* const* d_in, const int* in_sizes, int n_in,
                              void* d_out, int out_size, void* d_ws, size_t ws_size,
                              hipStream_t stream) {
    (void)in_sizes; (void)n_in; (void)out_size; (void)ws_size;

    const float* x  = (const float*)d_in[0];
    const float* Wq = (const float*)d_in[1];
    const float* bq = (const float*)d_in[2];
    const float* Wk = (const float*)d_in[3];
    const float* bk = (const float*)d_in[4];
    const float* Wv = (const float*)d_in[5];
    const float* bv = (const float*)d_in[6];

    const size_t SZ = (size_t)SEQ * HID;     // 16M elements
    float*     scores = (float*)d_out;       // S*S fp32 (64 MB)
    _Float16*  xh   = (_Float16*)d_ws;       // 32 MB
    _Float16*  wbuf = xh + SZ;               // 32 MB (staged weight, reused)
    _Float16*  qh   = wbuf + SZ;             // 32 MB
    _Float16*  kh   = qh + SZ;               // 32 MB
    float*     v    = (float*)(kh + SZ);     // 64 MB  -> 192 MB total

    const dim3 blk(256);
    const dim3 cgrid((unsigned)(SZ / 4 / 256));         // 16384 blocks
    const dim3 ggrid(HID / BN, SEQ / BM);               // 16 x 32

    cvt_f32_to_f16<<<cgrid, blk, 0, stream>>>(x, xh);

    cvt_f32_to_f16<<<cgrid, blk, 0, stream>>>(Wq, wbuf);
    gemm_nt_wmma<true, true><<<ggrid, blk, 0, stream>>>(xh, wbuf, bq, qh,
                                                        SEQ, HID, HID, 1.0f);
    cvt_f32_to_f16<<<cgrid, blk, 0, stream>>>(Wk, wbuf);
    gemm_nt_wmma<true, true><<<ggrid, blk, 0, stream>>>(xh, wbuf, bk, kh,
                                                        SEQ, HID, HID, 1.0f);
    cvt_f32_to_f16<<<cgrid, blk, 0, stream>>>(Wv, wbuf);
    gemm_nt_wmma<true, false><<<ggrid, blk, 0, stream>>>(xh, wbuf, bv, v,
                                                         SEQ, HID, HID, 1.0f);

    // scores = (q @ k^T) / sqrt(H) -> d_out
    gemm_nt_wmma<false, false><<<ggrid, blk, 0, stream>>>(qh, kh, nullptr, scores,
                                                          SEQ, SEQ, HID, 0.015625f);

    // out = softmax(scores) * v -> in place in d_out
    softmax_mix<<<SEQ, blk, 0, stream>>>(scores, v, SEQ);
}
